// SoftMoE_52252572123844
// MI455X (gfx1250) — matrix operand order repaired and checked by reading, not compiled, not run on hardware
//
#include <hip/hip_runtime.h>

// ---------------------------------------------------------------------------
// SoftMoE forward on gfx1250 (MI455X). bf16 WMMA (v_wmma_f32_16x16x32_bf16),
// f32 accumulate, wave32, 256-thread (8-wave) blocks.
// Double-buffered LDS: A tiles via GLOBAL_LOAD_ASYNC_TO_LDS_B128 (ASYNCcnt),
// B tiles staged pre-transposed (global->VGPR early, b64 scatter after
// compute), so every fragment read is 2x ds_load_b128 and global latency is
// hidden behind the WMMA stack.
// ---------------------------------------------------------------------------

#define TT   64
#define ZZ   576
#define DD   1024
#define NE   16
#define PS   64
#define HH   4096
#define TOK  (TT * ZZ)        // 36864 tokens
#define SL   (NE * PS)        // 1024 slots (== DD)
#define KSPLIT 4
#define KCH  (TOK / KSPLIT)   // 9216 tokens per split-K chunk

typedef __attribute__((ext_vector_type(16))) __bf16 v16bf;
typedef __attribute__((ext_vector_type(8)))  float  v8f;

__device__ __forceinline__ unsigned short f2bf(float f) {
  unsigned int u = __float_as_uint(f);
  u += 0x7FFFu + ((u >> 16) & 1u);        // round-to-nearest-even
  return (unsigned short)(u >> 16);
}

__device__ __forceinline__ v8f wmma_bf16(v16bf a, v16bf b, v8f c) {
  return __builtin_amdgcn_wmma_f32_16x16x32_bf16(
      false, a, false, b, (short)0, c, false, false);
}

// --- CDNA5 async global->LDS copy (16 B per lane, tracked by ASYNCcnt) ------
__device__ __forceinline__ void async_copy_b128(unsigned lds_byte_off,
                                                const void* gaddr) {
  asm volatile("global_load_async_to_lds_b128 %0, %1, off"
               :: "v"(lds_byte_off), "v"((unsigned long long)(size_t)gaddr)
               : "memory");
}
__device__ __forceinline__ void wait_async() {
  asm volatile("s_wait_asynccnt 0x0" ::: "memory");
}

// --- staging helpers --------------------------------------------------------
// Async direct copy: global row-major [M x 64] bf16 tile -> LDS as[m][k].
template <int M, int PA>
__device__ __forceinline__ void stage_direct_async(const unsigned short* __restrict__ g,
                                                   size_t gpitch,
                                                   unsigned short* __restrict__ as,
                                                   int tid) {
  constexpr int IT = (M * 8) / 256;   // units of 8 halves (16 B)
#pragma unroll
  for (int it = 0; it < IT; ++it) {
    const int u = tid + it * 256;
    const int m = u >> 3, k = (u & 7) * 8;
    async_copy_b128((unsigned)(size_t)(as + m * PA + k),
                    (const void*)(g + (size_t)m * gpitch + k));
  }
}

// Split transpose stage: load() pulls the [64(K) x C] panel into registers
// (uint4, coalesced along C); store() scatters it transposed to LDS ct[c][k]
// as packed b64 (lowered to ds_store_2addr_b64 pairs).
template <int C, int PB>
struct TStage {
  static constexpr int UN = C / 8;
  static constexpr int UNITS = 16 * UN;          // (64/4) k-units * c-units
  static constexpr int IT = (UNITS + 255) / 256;
  uint4 r[IT][4];

  __device__ __forceinline__ void load(const unsigned short* __restrict__ g,
                                       size_t gpitch, int tid) {
#pragma unroll
    for (int it = 0; it < IT; ++it) {
      const int u = tid + it * 256;
      if (UNITS % 256 != 0 && u >= UNITS) continue;
      const int c0 = (u % UN) * 8, k0 = (u / UN) * 4;
#pragma unroll
      for (int j = 0; j < 4; ++j)
        r[it][j] = *(const uint4*)(g + (size_t)(k0 + j) * gpitch + c0);
    }
  }
  __device__ __forceinline__ void store(unsigned short* __restrict__ ct, int tid) {
#pragma unroll
    for (int it = 0; it < IT; ++it) {
      const int u = tid + it * 256;
      if (UNITS % 256 != 0 && u >= UNITS) continue;
      const int c0 = (u % UN) * 8, k0 = (u / UN) * 4;
      union { uint4 v; unsigned short h[8]; } q[4];
#pragma unroll
      for (int j = 0; j < 4; ++j) q[j].v = r[it][j];
#pragma unroll
      for (int c = 0; c < 8; ++c) {
        unsigned long long w = (unsigned long long)q[0].h[c]
                             | ((unsigned long long)q[1].h[c] << 16)
                             | ((unsigned long long)q[2].h[c] << 32)
                             | ((unsigned long long)q[3].h[c] << 48);
        *(unsigned long long*)(ct + (size_t)(c0 + c) * PB + k0) = w;
      }
    }
  }
};

// --- fragment loads (each lowers to 2x ds_load_b128) ------------------------
__device__ __forceinline__ v16bf load_frag_a(const unsigned short* as, int pitch,
                                             int m0, int k0, int lane) {
  const int m = m0 + (lane & 15);
  const int kb = (lane >> 4) * 16;
  union { uint4 q[2]; v16bf v; } u;
  u.q[0] = *(const uint4*)(as + m * pitch + k0 + kb);
  u.q[1] = *(const uint4*)(as + m * pitch + k0 + kb + 8);
  return u.v;
}
__device__ __forceinline__ v16bf load_frag_b(const unsigned short* bt, int pitch,
                                             int k0, int n0, int lane) {
  const int n = n0 + (lane & 15);
  const int kb = k0 + (lane >> 4) * 16;
  union { uint4 q[2]; v16bf v; } u;
  u.q[0] = *(const uint4*)(bt + n * pitch + kb);
  u.q[1] = *(const uint4*)(bt + n * pitch + kb + 8);
  return u.v;
}

// ---------------------------------------------------------------------------
// Kernel 0: vectorized f32 -> bf16 (float4 in, packed u64 out)
// ---------------------------------------------------------------------------
__global__ __launch_bounds__(256) void k_cvt_bf16(const float4* __restrict__ src,
                                                  unsigned long long* __restrict__ dst,
                                                  size_t n4) {
  size_t i = (size_t)blockIdx.x * blockDim.x + threadIdx.x;
  const size_t stride = (size_t)gridDim.x * blockDim.x;
  for (; i < n4; i += stride) {
    const float4 f = src[i];
    dst[i] = (unsigned long long)f2bf(f.x)
           | ((unsigned long long)f2bf(f.y) << 16)
           | ((unsigned long long)f2bf(f.z) << 32)
           | ((unsigned long long)f2bf(f.w) << 48);
  }
}

// ---------------------------------------------------------------------------
// Kernel 1: logits = x @ phi_w + phi_b (32 tokens x 1024 slots per WG), logits
// tile in LDS, then dispatch softmax (over P) + combine softmax (over N).
// ---------------------------------------------------------------------------
__global__ __launch_bounds__(256) void k_logits_softmax(
    const unsigned short* __restrict__ xbf, const unsigned short* __restrict__ phibf,
    const float* __restrict__ phi_b,
    unsigned short* __restrict__ Dbf, unsigned short* __restrict__ Cbf) {
  __shared__ unsigned short as[32 * 72];
  __shared__ unsigned short bt[256 * 72];
  __shared__ float lg[32 * 1024];           // 128 KB

  const int tid = threadIdx.x, lane = tid & 31, wv = tid >> 5;
  const int tok0 = blockIdx.x * 32;
  const int wm = wv >> 2;          // 0..1
  const int wn4 = (wv & 3) * 4;
  TStage<256, 72> bs;

  for (int np = 0; np < 4; ++np) {
    v8f acc[4] = {};
    for (int k0 = 0; k0 < DD; k0 += 64) {
      __syncthreads();
      stage_direct_async<32, 72>(xbf + (size_t)tok0 * DD + k0, DD, as, tid);
      bs.load(phibf + (size_t)k0 * SL + np * 256, SL, tid);
      bs.store(bt, tid);
      wait_async();
      __syncthreads();
#pragma unroll
      for (int ks = 0; ks < 64; ks += 32) {
        const v16bf a = load_frag_a(as, 72, wm * 16, ks, lane);
#pragma unroll
        for (int j = 0; j < 4; ++j) {
          const v16bf b = load_frag_b(bt, 72, ks, (wn4 + j) * 16, lane);
          acc[j] = wmma_bf16(a, b, acc[j]);
        }
      }
    }
    __syncthreads();
#pragma unroll
    for (int j = 0; j < 4; ++j) {
      const int col = np * 256 + (wn4 + j) * 16 + (lane & 15);
      const float bias = phi_b[col];
#pragma unroll
      for (int r = 0; r < 8; ++r) {
        const int row = wm * 16 + r + (lane >> 4) * 8;
        lg[row * 1024 + col] = acc[j][r] + bias;
      }
    }
  }
  __syncthreads();

  for (int g = tid; g < 32 * 16; g += 256) {        // dispatch softmax
    const int row = g >> 4, e = g & 15;
    const float* p = &lg[row * 1024 + e * 64];
    float mx = -3.4e38f;
    for (int j = 0; j < 64; ++j) mx = fmaxf(mx, p[j]);
    float s = 0.f;
    for (int j = 0; j < 64; ++j) s += __expf(p[j] - mx);
    const float inv = 1.0f / s;
    const size_t base = (size_t)(tok0 + row) * SL + e * 64;
    for (int j = 0; j < 64; ++j) Dbf[base + j] = f2bf(__expf(p[j] - mx) * inv);
  }
  for (int g = tid; g < 32 * 64; g += 256) {        // combine softmax
    const int row = g >> 6, pp = g & 63;
    const float* p = &lg[row * 1024 + pp];
    float mx = -3.4e38f;
    for (int n = 0; n < 16; ++n) mx = fmaxf(mx, p[n * 64]);
    float s = 0.f;
    for (int n = 0; n < 16; ++n) s += __expf(p[n * 64] - mx);
    const float inv = 1.0f / s;
    const size_t base = (size_t)(tok0 + row) * SL + pp;
    for (int n = 0; n < 16; ++n)
      Cbf[base + (size_t)n * 64] = f2bf(__expf(p[n * 64] - mx) * inv);
  }
}

// ---------------------------------------------------------------------------
// Kernel 2: X_tilde partials = D^T @ x over a split-K token chunk.
// 64x64 tile per WG, grid (256, KSPLIT). Both operands K-major -> transposed.
// Double-buffered LDS, one barrier per k-tile.
// ---------------------------------------------------------------------------
__global__ __launch_bounds__(256) void k_dispatch_mix(
    const unsigned short* __restrict__ Dbf, const unsigned short* __restrict__ xbf,
    float* __restrict__ Xtp) {
  __shared__ unsigned short at[2][64 * 72];
  __shared__ unsigned short bt[2][64 * 72];
  const int tid = threadIdx.x, lane = tid & 31, wv = tid >> 5;
  const int s0 = (blockIdx.x & 15) * 64;
  const int d0 = (blockIdx.x >> 4) * 64;
  const int kc = blockIdx.y;
  const int wm = wv >> 1, wn2 = (wv & 1) * 2;
  v8f acc[2] = {};
  TStage<64, 72> sa, sb;

  const int kbeg = kc * KCH, kend = kbeg + KCH;
  // prologue: stage tile 0 into buffer 0
  sa.load(Dbf + (size_t)kbeg * SL + s0, SL, tid);
  sb.load(xbf + (size_t)kbeg * DD + d0, DD, tid);
  sa.store(at[0], tid);
  sb.store(bt[0], tid);
  __syncthreads();

  for (int k0 = kbeg; k0 < kend; k0 += 64) {
    const int cur = (k0 >> 6) & 1, nxt = cur ^ 1;
    const bool more = (k0 + 64) < kend;
    if (more) {
      sa.load(Dbf + (size_t)(k0 + 64) * SL + s0, SL, tid);
      sb.load(xbf + (size_t)(k0 + 64) * DD + d0, DD, tid);
      if (k0 + 128 < kend) {   // warm L2 two tiles ahead
        __builtin_prefetch(&Dbf[(size_t)(k0 + 128 + (tid >> 3)) * SL + s0], 0, 1);
        __builtin_prefetch(&xbf[(size_t)(k0 + 128 + (tid >> 3)) * DD + d0], 0, 1);
      }
    }
#pragma unroll
    for (int ks = 0; ks < 64; ks += 32) {
      const v16bf a = load_frag_a(at[cur], 72, wm * 16, ks, lane);
#pragma unroll
      for (int j = 0; j < 2; ++j) {
        const v16bf b = load_frag_b(bt[cur], 72, ks, (wn2 + j) * 16, lane);
        acc[j] = wmma_bf16(a, b, acc[j]);
      }
    }
    if (more) { sa.store(at[nxt], tid); sb.store(bt[nxt], tid); }
    __syncthreads();
  }
  float* out = Xtp + (size_t)kc * SL * DD;
#pragma unroll
  for (int j = 0; j < 2; ++j) {
    const int col = d0 + (wn2 + j) * 16 + (lane & 15);
#pragma unroll
    for (int r = 0; r < 8; ++r) {
      const int row = s0 + wm * 16 + r + (lane >> 4) * 8;
      out[(size_t)row * DD + col] = acc[j][r];   // raw partial; clip in reduce
    }
  }
}

// ---------------------------------------------------------------------------
// Kernel 3: reduce split-K partials, clip, LayerNorm (no affine), tanh -> bf16
// ---------------------------------------------------------------------------
__global__ __launch_bounds__(256) void k_ln_tanh(const float* __restrict__ Xtp,
                                                 unsigned short* __restrict__ Xtb) {
  __shared__ float rowv[1024];
  __shared__ float rs[256], rss[256];
  const int row = blockIdx.x, tid = threadIdx.x;
  float s = 0.f, ss = 0.f;
  for (int i = tid; i < DD; i += 256) {
    float v = 0.f;
#pragma unroll
    for (int c = 0; c < KSPLIT; ++c)
      v += Xtp[(size_t)c * SL * DD + (size_t)row * DD + i];
    v = fminf(fmaxf(v, -33000.0f), 65000.0f);
    rowv[i] = v; s += v; ss += v * v;
  }
  rs[tid] = s; rss[tid] = ss;
  __syncthreads();
  for (int off = 128; off; off >>= 1) {
    if (tid < off) { rs[tid] += rs[tid + off]; rss[tid] += rss[tid + off]; }
    __syncthreads();
  }
  const float mu  = rs[0] * (1.0f / DD);
  const float var = rss[0] * (1.0f / DD) - mu * mu;
  const float inv = rsqrtf(var + 1e-5f);
  for (int i = tid; i < DD; i += 256)
    Xtb[(size_t)row * DD + i] = f2bf(tanhf((rowv[i] - mu) * inv));
}

// ---------------------------------------------------------------------------
// Kernel 4: per-expert GEMM Y_tilde[n] = Xtb[n] (64x1024) @ W[n] (1024x4096)
// + b[n]. WG tile 64x256 (2 M-tiles x 4 N-tiles per wave), double-buffered.
// ---------------------------------------------------------------------------
__global__ __launch_bounds__(256) void k_expert_gemm(
    const unsigned short* __restrict__ Xtb, const unsigned short* __restrict__ ewb,
    const float* __restrict__ eb, unsigned short* __restrict__ Ytb) {
  __shared__ unsigned short as[2][64 * 72];
  __shared__ unsigned short bt[2][256 * 72];
  const int tid = threadIdx.x, lane = tid & 31, wv = tid >> 5;
  const int h0 = blockIdx.x * 256;
  const int ex = blockIdx.y;
  const unsigned short* A = Xtb + (size_t)ex * PS * DD;
  const unsigned short* B = ewb + (size_t)ex * DD * HH;
  const int wmp = (wv >> 2) * 2;
  const int wn4 = (wv & 3) * 4;
  v8f acc[2][4] = {};
  TStage<256, 72> sb;

  stage_direct_async<64, 72>(A, DD, as[0], tid);
  sb.load(B + h0, HH, tid);
  sb.store(bt[0], tid);
  wait_async();
  __syncthreads();

  for (int k0 = 0; k0 < DD; k0 += 64) {
    const int cur = (k0 >> 6) & 1, nxt = cur ^ 1;
    const bool more = (k0 + 64) < DD;
    if (more) {
      stage_direct_async<64, 72>(A + k0 + 64, DD, as[nxt], tid);
      sb.load(B + (size_t)(k0 + 64) * HH + h0, HH, tid);
    }
#pragma unroll
    for (int ks = 0; ks < 64; ks += 32) {
      const v16bf a0 = load_frag_a(as[cur], 72, (wmp + 0) * 16, ks, lane);
      const v16bf a1 = load_frag_a(as[cur], 72, (wmp + 1) * 16, ks, lane);
#pragma unroll
      for (int j = 0; j < 4; ++j) {
        const v16bf b = load_frag_b(bt[cur], 72, ks, (wn4 + j) * 16, lane);
        acc[0][j] = wmma_bf16(a0, b, acc[0][j]);
        acc[1][j] = wmma_bf16(a1, b, acc[1][j]);
      }
    }
    if (more) sb.store(bt[nxt], tid);
    wait_async();
    __syncthreads();
  }
#pragma unroll
  for (int i = 0; i < 2; ++i)
#pragma unroll
    for (int j = 0; j < 4; ++j) {
      const int col = h0 + (wn4 + j) * 16 + (lane & 15);
      const float bias = eb[ex * HH + col];
#pragma unroll
      for (int r = 0; r < 8; ++r) {
        const int row = (wmp + i) * 16 + r + (lane >> 4) * 8;
        Ytb[((size_t)ex * PS + row) * HH + col] = f2bf(acc[i][j][r] + bias);
      }
    }
}

// ---------------------------------------------------------------------------
// Kernel 5: Y = C (36864x1024) @ Y_tilde (1024x4096), f32 out.
// WG tile 128x128 (2 M-tiles x 4 N-tiles per wave), double-buffered.
// ---------------------------------------------------------------------------
__global__ __launch_bounds__(256) void k_combine(
    const unsigned short* __restrict__ Cbf, const unsigned short* __restrict__ Ytb,
    float* __restrict__ Y) {
  __shared__ unsigned short as[2][128 * 72];
  __shared__ unsigned short bt[2][128 * 72];
  const int tid = threadIdx.x, lane = tid & 31, wv = tid >> 5;
  const int t0 = blockIdx.y * 128;
  const int h0 = blockIdx.x * 128;
  const int wmp = (wv >> 1) * 2;
  const int wn4 = (wv & 1) * 4;
  v8f acc[2][4] = {};
  TStage<128, 72> sb;

  stage_direct_async<128, 72>(Cbf + (size_t)t0 * SL, SL, as[0], tid);
  sb.load(Ytb + h0, HH, tid);
  sb.store(bt[0], tid);
  wait_async();
  __syncthreads();

  for (int k0 = 0; k0 < SL; k0 += 64) {
    const int cur = (k0 >> 6) & 1, nxt = cur ^ 1;
    const bool more = (k0 + 64) < SL;
    if (more) {
      stage_direct_async<128, 72>(Cbf + (size_t)t0 * SL + k0 + 64, SL, as[nxt], tid);
      sb.load(Ytb + (size_t)(k0 + 64) * HH + h0, HH, tid);
    }
#pragma unroll
    for (int ks = 0; ks < 64; ks += 32) {
      const v16bf a0 = load_frag_a(as[cur], 72, (wmp + 0) * 16, ks, lane);
      const v16bf a1 = load_frag_a(as[cur], 72, (wmp + 1) * 16, ks, lane);
#pragma unroll
      for (int j = 0; j < 4; ++j) {
        const v16bf b = load_frag_b(bt[cur], 72, ks, (wn4 + j) * 16, lane);
        acc[0][j] = wmma_bf16(a0, b, acc[0][j]);
        acc[1][j] = wmma_bf16(a1, b, acc[1][j]);
      }
    }
    if (more) sb.store(bt[nxt], tid);
    wait_async();
    __syncthreads();
  }
#pragma unroll
  for (int i = 0; i < 2; ++i)
#pragma unroll
    for (int j = 0; j < 4; ++j) {
      const int col = h0 + (wn4 + j) * 16 + (lane & 15);
#pragma unroll
      for (int r = 0; r < 8; ++r) {
        const int row = t0 + (wmp + i) * 16 + r + (lane >> 4) * 8;
        Y[(size_t)row * HH + col] = acc[i][j][r];
      }
    }
}

// ---------------------------------------------------------------------------
extern "C" void kernel_launch(void* const* d_in, const int* in_sizes, int n_in,
                              void* d_out, int out_size, void* d_ws, size_t ws_size,
                              hipStream_t stream) {
  (void)in_sizes; (void)n_in; (void)out_size; (void)ws_size;
  const float* x        = (const float*)d_in[0];
  const float* phi_w    = (const float*)d_in[1];
  const float* phi_b    = (const float*)d_in[2];
  const float* expert_w = (const float*)d_in[3];
  const float* expert_b = (const float*)d_in[4];
  float* Y = (float*)d_out;

  char* ws = (char*)d_ws;
  size_t off = 0;
  auto carve = [&](size_t bytes) {
    void* p = ws + off;
    off = (off + bytes + 255) & ~(size_t)255;
    return p;
  };
  unsigned short* xbf   = (unsigned short*)carve((size_t)TOK * DD * 2);      // 75.5 MB
  unsigned short* Dbf   = (unsigned short*)carve((size_t)TOK * SL * 2);      // 75.5 MB
  unsigned short* Cbf   = (unsigned short*)carve((size_t)TOK * SL * 2);      // 75.5 MB
  unsigned short* ewb   = (unsigned short*)carve((size_t)NE * DD * HH * 2);  // 134 MB
  unsigned short* phibf = (unsigned short*)carve((size_t)DD * SL * 2);       // 2 MB
  float*          Xtp   = (float*)carve((size_t)KSPLIT * SL * DD * 4);       // 16 MB
  unsigned short* Xtb   = (unsigned short*)carve((size_t)SL * DD * 2);       // 2 MB
  unsigned short* Ytb   = (unsigned short*)carve((size_t)NE * PS * HH * 2);  // 8 MB

  k_cvt_bf16<<<2048, 256, 0, stream>>>((const float4*)x, (unsigned long long*)xbf,
                                       (size_t)TOK * DD / 4);
  k_cvt_bf16<<<4096, 256, 0, stream>>>((const float4*)expert_w, (unsigned long long*)ewb,
                                       (size_t)NE * DD * HH / 4);
  k_cvt_bf16<<<512, 256, 0, stream>>>((const float4*)phi_w, (unsigned long long*)phibf,
                                      (size_t)DD * SL / 4);
  k_logits_softmax<<<TOK / 32, 256, 0, stream>>>(xbf, phibf, phi_b, Dbf, Cbf);
  k_dispatch_mix<<<dim3(256, KSPLIT), 256, 0, stream>>>(Dbf, xbf, Xtp);
  k_ln_tanh<<<SL, 256, 0, stream>>>(Xtp, Xtb);
  k_expert_gemm<<<dim3(HH / 256, NE), 256, 0, stream>>>(Xtb, ewb, expert_b, Ytb);
  k_combine<<<dim3(HH / 128, TOK / 128), 256, 0, stream>>>(Cbf, Ytb, Y);
}
